// Decoder_12309376270874
// MI455X (gfx1250) — matrix operand order repaired and checked by reading, not compiled, not run on hardware
//
#include <hip/hip_runtime.h>
#include <hip/hip_bf16.h>

// ---------------------------------------------------------------------------
// Transformer decoder forward (B=1, S=2048, D=512, H=8, L=4, DFF=2048, V=32000)
// MI455X / gfx1250: wave32, WMMA f32_16x16x32_f16, LDS-staged GEMM tiles
// (async GLOBAL_LOAD_ASYNC_TO_LDS_B128 when the toolchain exposes it),
// fused flash-attention (no S x S score materialization), V pre-transposed so
// every WMMA operand load is a contiguous b128.
// ---------------------------------------------------------------------------

#define D_    512
#define H_    8
#define DK_   64
#define L_    4
#define V_    32000
#define S_    2048
#define DFF_  2048
#define EPS_  1.1920929e-07f

typedef _Float16 v16h __attribute__((ext_vector_type(16)));
typedef _Float16 v8h  __attribute__((ext_vector_type(8)));
typedef float    v8f  __attribute__((ext_vector_type(8)));
typedef int      v4i  __attribute__((ext_vector_type(4)));

#if defined(__has_builtin)
#  if __has_builtin(__builtin_amdgcn_global_load_async_to_lds_b128)
#    define USE_ASYNC_LDS 1
#  endif
#endif

#ifdef USE_ASYNC_LDS
// signature (from clang diagnostic): (v4i AS1*, v4i AS3*, imm offset, imm cpol)
__device__ __forceinline__ void async_copy_b128(const void* g, void* l) {
  __builtin_amdgcn_global_load_async_to_lds_b128(
      (__attribute__((address_space(1))) v4i*)g,
      (__attribute__((address_space(3))) v4i*)l, 0, 0);
}
__device__ __forceinline__ void async_wait() {
#  if __has_builtin(__builtin_amdgcn_s_wait_asynccnt)
  __builtin_amdgcn_s_wait_asynccnt(0);
#  else
  asm volatile("s_wait_asynccnt 0" ::: "memory");
#  endif
}
#endif

__device__ __forceinline__ v8f wmma_f16(v16h a, v16h b, v8f c) {
  // D = A(16x32 f16) x B(32x16 f16) + C(16x16 f32)
  return __builtin_amdgcn_wmma_f32_16x16x32_f16(
      /*neg_a=*/false, a, /*neg_b=*/false, b,
      /*c_mod=*/(short)0, c, /*reuse_a=*/false, /*reuse_b=*/false);
}

__device__ __forceinline__ v16h cat8(v8h lo, v8h hi) {
  return __builtin_shufflevector(lo, hi, 0,1,2,3,4,5,6,7,8,9,10,11,12,13,14,15);
}

// ---------------------------------------------------------------------------
// fp32 -> f16 conversion (weights)
// ---------------------------------------------------------------------------
__global__ void cvt_f32_f16_kernel(const float* __restrict__ in,
                                   _Float16* __restrict__ out, long long n) {
  long long i = (long long)blockIdx.x * blockDim.x + threadIdx.x;
  long long stride = (long long)gridDim.x * blockDim.x;
  for (; i < n; i += stride) out[i] = (_Float16)in[i];
}

// ---------------------------------------------------------------------------
// f16 transpose: in [S_, D_] -> out [D_, S_]  (LDS 32x32 tiles, +1 pad)
// grid = (D_/32, S_/32), block = 256 (32x8)
// ---------------------------------------------------------------------------
__global__ __launch_bounds__(256)
void transpose_f16_kernel(const _Float16* __restrict__ in,
                          _Float16* __restrict__ outT) {
  __shared__ _Float16 tile[32][33];
  const int bx = blockIdx.x;          // along D
  const int by = blockIdx.y;          // along S
  const int tx = threadIdx.x & 31;
  const int ty = threadIdx.x >> 5;    // 8 rows per pass
#pragma unroll
  for (int i = 0; i < 32; i += 8)
    tile[ty + i][tx] = in[(size_t)(by * 32 + ty + i) * D_ + bx * 32 + tx];
  __syncthreads();
#pragma unroll
  for (int i = 0; i < 32; i += 8)
    outT[(size_t)(bx * 32 + ty + i) * S_ + by * 32 + tx] = tile[tx][ty + i];
}

// ---------------------------------------------------------------------------
// Embedding + sinusoidal positional encoding: x fp32 + f16 copy
// grid = S, block = 256 (thread j owns dims 2j, 2j+1)
// ---------------------------------------------------------------------------
__global__ __launch_bounds__(256)
void embed_kernel(const int* __restrict__ ids, const float* __restrict__ emb,
                  float* __restrict__ xo, _Float16* __restrict__ xh) {
  const int s = blockIdx.x;
  const int j = threadIdx.x;                   // 0..255 -> frequency index
  const int id = ids[s];
  const float inv = powf(10000.0f, (2.0f * (float)j) / (float)D_);
  const float ang = (float)s / inv;
  const size_t o = (size_t)s * D_ + 2 * j;
  const float e0 = emb[(size_t)id * D_ + 2 * j]     + sinf(ang);
  const float e1 = emb[(size_t)id * D_ + 2 * j + 1] + cosf(ang);
  xo[o] = e0; xo[o + 1] = e1;
  xh[o] = (_Float16)e0; xh[o + 1] = (_Float16)e1;
}

// ---------------------------------------------------------------------------
// Generic WMMA GEMM: out = act(A[M,K] * W[N,K]^T + bias [+ res]) (row-major)
// Block 256 threads = 8 waves; 128x64 output tile, K-step 32 through LDS.
// Wave (wm in 0..3, wn in 0..1) owns a 32x32 sub-tile: 2x2 accumulators,
// 4 WMMAs per k-step, each A/B fragment reused twice.
// ---------------------------------------------------------------------------
template <int GELU>
__global__ __launch_bounds__(256)
void gemm_f16_kernel(const _Float16* __restrict__ A, const _Float16* __restrict__ W,
                     const float* __restrict__ bias, const float* __restrict__ res,
                     float* __restrict__ outF, _Float16* __restrict__ outH,
                     int M, int N, int K) {
  __shared__ __align__(32) _Float16 As[128 * 32];
  __shared__ __align__(32) _Float16 Bs[64 * 32];

  const int tid   = threadIdx.x;
  const int nbase = blockIdx.x * 64;
  const int mbase = blockIdx.y * 128;
  const int wave  = tid >> 5;
  const int lane  = tid & 31;
  const int wm    = wave >> 1;       // 0..3 -> 32 rows each
  const int wn    = wave & 1;        // 0..1 -> 32 cols each
  const int half  = lane >> 4;
  const int l15   = lane & 15;

  const int arow0 = tid >> 2,  avec0 = tid & 3;          // A chunk 0 (rows 0..63)
  const int arow1 = arow0 + 64;                          // A chunk 1 (rows 64..127)

  v8f acc[2][2] = {};

  const int ktiles = K >> 5;
  for (int kt = 0; kt < ktiles; ++kt) {
    const int k0 = kt << 5;
    // stage 128x32 A tile and 64x32 W tile
#ifdef USE_ASYNC_LDS
    async_copy_b128(&A[(size_t)(mbase + arow0) * K + k0 + avec0 * 8],
                    &As[arow0 * 32 + avec0 * 8]);
    async_copy_b128(&A[(size_t)(mbase + arow1) * K + k0 + avec0 * 8],
                    &As[arow1 * 32 + avec0 * 8]);
    async_copy_b128(&W[(size_t)(nbase + arow0) * K + k0 + avec0 * 8],
                    &Bs[arow0 * 32 + avec0 * 8]);
#else
    *(v8h*)&As[arow0 * 32 + avec0 * 8] =
        *(const v8h*)&A[(size_t)(mbase + arow0) * K + k0 + avec0 * 8];
    *(v8h*)&As[arow1 * 32 + avec0 * 8] =
        *(const v8h*)&A[(size_t)(mbase + arow1) * K + k0 + avec0 * 8];
    *(v8h*)&Bs[arow0 * 32 + avec0 * 8] =
        *(const v8h*)&W[(size_t)(nbase + arow0) * K + k0 + avec0 * 8];
#endif
    if (kt + 1 < ktiles) {
      __builtin_prefetch(&A[(size_t)(mbase + arow0) * K + k0 + 32], 0, 3);
      __builtin_prefetch(&W[(size_t)(nbase + arow0) * K + k0 + 32], 0, 3);
    }
#ifdef USE_ASYNC_LDS
    async_wait();
#endif
    __syncthreads();

    // A fragments (16x32): lane row; k = e + 16*half contiguous
    v16h af[2];
#pragma unroll
    for (int sm = 0; sm < 2; ++sm) {
      const int arow = wm * 32 + sm * 16 + l15;
      af[sm] = *(const v16h*)&As[arow * 32 + 16 * half];
    }
    // B fragments (32x16): column n = l15; k = e + 16*half contiguous
    v16h bf[2];
#pragma unroll
    for (int sn = 0; sn < 2; ++sn)
      bf[sn] = *(const v16h*)&Bs[(wn * 32 + sn * 16 + l15) * 32 + 16 * half];

#pragma unroll
    for (int sm = 0; sm < 2; ++sm)
#pragma unroll
      for (int sn = 0; sn < 2; ++sn)
        acc[sm][sn] = wmma_f16(af[sm], bf[sn], acc[sm][sn]);
    __syncthreads();
  }

  // Epilogue: C/D layout -> VGPR r holds (m = r + 8*half, n = l15)
#pragma unroll
  for (int sm = 0; sm < 2; ++sm)
#pragma unroll
    for (int sn = 0; sn < 2; ++sn)
#pragma unroll
      for (int r = 0; r < 8; ++r) {
        const int grow = mbase + wm * 32 + sm * 16 + r + 8 * half;
        const int gcol = nbase + wn * 32 + sn * 16 + l15;
        float v = acc[sm][sn][r] + bias[gcol];
        if (res) v += res[(size_t)grow * N + gcol];
        if (GELU) v = 0.5f * v * (1.0f + erff(v * 0.70710678118654752f));
        if (outF) outF[(size_t)grow * N + gcol] = v;
        if (outH) outH[(size_t)grow * N + gcol] = (_Float16)v;
      }
}

// ---------------------------------------------------------------------------
// Fused flash attention, faithful to reference semantics:
//   scores[k,q] = K[k] . Q[q];  softmax over q with mask (q <= k && amask[q]);
//   out[k] = sum_q p[k,q] * V[q]
// One wave per (16 k-rows, head). Online softmax, P re-layout via LDS.
// Vt is V transposed to [D, S] so the P@V B-operand loads are contiguous b128.
// ---------------------------------------------------------------------------
__global__ __launch_bounds__(32)
void flash_attn_kernel(const _Float16* __restrict__ Qh, const _Float16* __restrict__ Kh,
                       const _Float16* __restrict__ Vt, const int* __restrict__ amask,
                       _Float16* __restrict__ outH) {
  __shared__ __align__(32) _Float16 P[16 * 32];

  const int lane  = threadIdx.x & 31;
  const int half  = lane >> 4;
  const int l15   = lane & 15;
  const int kbase = blockIdx.x * 16;
  const int hoff  = blockIdx.y * DK_;

  // A operand: K rows for this block, DK=64 split in two 32-wide chunks
  const _Float16* krow = &Kh[(size_t)(kbase + l15) * D_ + hoff];
  v16h ka[2];
#pragma unroll
  for (int ch = 0; ch < 2; ++ch) {
    ka[ch] = *(const v16h*)&krow[ch * 32 + 16 * half];
  }

  float mrun[8], lrun[8];
  v8f acc[4] = {};
#pragma unroll
  for (int r = 0; r < 8; ++r) { mrun[r] = -1e38f; lrun[r] = 0.0f; }

  const int kend = kbase + 15;
  for (int qb = 0; qb <= kend; qb += 32) {
    // scores: two 16x16 tiles (q tiles), sum over d via 2 WMMAs each
    v8f s0 = {}, s1 = {};
#pragma unroll
    for (int ch = 0; ch < 2; ++ch) {
      // B element (d, q) = Q[q, d]: contiguous 16 f16 along d per lane/column
      v16h bq0 = *(const v16h*)&Qh[(size_t)(qb + l15) * D_ + hoff + ch * 32 + 16 * half];
      v16h bq1 = *(const v16h*)&Qh[(size_t)(qb + 16 + l15) * D_ + hoff + ch * 32 + 16 * half];
      s0 = wmma_f16(ka[ch], bq0, s0);
      s1 = wmma_f16(ka[ch], bq1, s1);
    }
    if (qb + 32 <= kend)
      __builtin_prefetch(&Qh[(size_t)(qb + 32 + l15) * D_ + hoff], 0, 3);

    const int am0 = amask[qb + l15];
    const int am1 = amask[qb + 16 + l15];
    const int q0  = qb + l15;
    const int q1  = qb + 16 + l15;

#pragma unroll
    for (int r = 0; r < 8; ++r) {
      const int krow_i = kbase + r + 8 * half;
      float v0 = ((q0 <= krow_i) && am0) ? s0[r] : -1e38f;
      float v1 = ((q1 <= krow_i) && am1) ? s1[r] : -1e38f;

      // row max across the 16 lanes holding this row's columns
      float rm = fmaxf(v0, v1);
      rm = fmaxf(rm, __shfl_xor(rm, 1));
      rm = fmaxf(rm, __shfl_xor(rm, 2));
      rm = fmaxf(rm, __shfl_xor(rm, 4));
      rm = fmaxf(rm, __shfl_xor(rm, 8));
      const float mnew = fmaxf(mrun[r], rm);

      float e0, e1, scale;
      if (mnew > -1e37f) {
        e0 = __expf(v0 - mnew);
        e1 = __expf(v1 - mnew);
        scale = __expf(mrun[r] - mnew);
      } else {
        e0 = 0.0f; e1 = 0.0f; scale = 1.0f;   // fully masked so far
      }
      float rs = e0 + e1;
      rs += __shfl_xor(rs, 1);
      rs += __shfl_xor(rs, 2);
      rs += __shfl_xor(rs, 4);
      rs += __shfl_xor(rs, 8);

      lrun[r] = lrun[r] * scale + rs;
      mrun[r] = mnew;
      acc[0][r] *= scale; acc[1][r] *= scale;
      acc[2][r] *= scale; acc[3][r] *= scale;

      // stash P in row-major 16x32 for C->A fragment re-layout
      P[(r + 8 * half) * 32 + l15]      = (_Float16)e0;
      P[(r + 8 * half) * 32 + 16 + l15] = (_Float16)e1;
    }
    __syncthreads();

    // P as A operand (16 rows k-block x 32 cols q)
    v16h pa = *(const v16h*)&P[l15 * 32 + 16 * half];

    // V as B operand from Vt[D, S]: B(qlocal, d) = Vt[hoff+d, qb+qlocal];
    // k = e + 16*half -> contiguous along S -> one v16h load per lane
#pragma unroll
    for (int dt = 0; dt < 4; ++dt) {
      v16h bv = *(const v16h*)&Vt[(size_t)(hoff + dt * 16 + l15) * S_ + qb + 16 * half];
      acc[dt] = wmma_f16(pa, bv, acc[dt]);
    }
    __syncthreads();
  }

  // normalize and store f16 attention output [S, D]
#pragma unroll
  for (int dt = 0; dt < 4; ++dt)
#pragma unroll
    for (int r = 0; r < 8; ++r) {
      const int row = kbase + r + 8 * half;
      const int col = hoff + dt * 16 + l15;
      const float denom = lrun[r];
      const float o = (denom > 0.0f) ? acc[dt][r] / denom : 0.0f;
      outH[(size_t)row * D_ + col] = (_Float16)o;
    }
}

// ---------------------------------------------------------------------------
// RMSNorm over D=512 per row; writes fp32 + f16. grid = S, block = 128.
// ---------------------------------------------------------------------------
__global__ __launch_bounds__(128)
void rmsnorm_kernel(const float* __restrict__ y, const float* __restrict__ g,
                    float* __restrict__ xo, _Float16* __restrict__ xh) {
  const int row = blockIdx.x;
  const int t   = threadIdx.x;
  __shared__ float red[4];

  float v[4];
  float ss = 0.0f;
#pragma unroll
  for (int i = 0; i < 4; ++i) {
    v[i] = y[(size_t)row * D_ + t * 4 + i];
    ss += v[i] * v[i];
  }
  ss += __shfl_xor(ss, 16);
  ss += __shfl_xor(ss, 8);
  ss += __shfl_xor(ss, 4);
  ss += __shfl_xor(ss, 2);
  ss += __shfl_xor(ss, 1);
  if ((t & 31) == 0) red[t >> 5] = ss;
  __syncthreads();
  const float tot  = red[0] + red[1] + red[2] + red[3];
  const float rinv = rsqrtf(tot * (1.0f / (float)D_) + EPS_);
#pragma unroll
  for (int i = 0; i < 4; ++i) {
    const float o = v[i] * rinv * g[t * 4 + i];
    xo[(size_t)row * D_ + t * 4 + i] = o;
    xh[(size_t)row * D_ + t * 4 + i] = (_Float16)o;
  }
}

// ---------------------------------------------------------------------------
// Host driver
// ---------------------------------------------------------------------------
extern "C" void kernel_launch(void* const* d_in, const int* in_sizes, int n_in,
                              void* d_out, int out_size, void* d_ws, size_t ws_size,
                              hipStream_t stream) {
  (void)in_sizes; (void)n_in; (void)out_size; (void)ws_size;

  const int*   ids    = (const int*)d_in[0];
  const int*   amask  = (const int*)d_in[1];
  const float* emb    = (const float*)d_in[2];
  const float* Wq     = (const float*)d_in[3];
  const float* bq     = (const float*)d_in[4];
  const float* Wk     = (const float*)d_in[5];
  const float* bk     = (const float*)d_in[6];
  const float* Wv     = (const float*)d_in[7];
  const float* bv     = (const float*)d_in[8];
  const float* Wo     = (const float*)d_in[9];
  const float* bo     = (const float*)d_in[10];
  const float* g1     = (const float*)d_in[11];
  const float* g2     = (const float*)d_in[12];
  const float* W1     = (const float*)d_in[13];
  const float* b1     = (const float*)d_in[14];
  const float* W2     = (const float*)d_in[15];
  const float* b2     = (const float*)d_in[16];
  const float* head_w = (const float*)d_in[17];
  const float* head_b = (const float*)d_in[18];
  float* out = (float*)d_out;

  // workspace carve (256B-aligned): ~56MB f16 weights + ~30MB activations
  char* p = (char*)d_ws;
  auto carve = [&](size_t bytes) -> void* {
    void* r = (void*)p;
    p += (bytes + 255) & ~(size_t)255;
    return r;
  };
  _Float16* WqH   = (_Float16*)carve((size_t)L_ * D_ * D_ * 2);
  _Float16* WkH   = (_Float16*)carve((size_t)L_ * D_ * D_ * 2);
  _Float16* WvH   = (_Float16*)carve((size_t)L_ * D_ * D_ * 2);
  _Float16* WoH   = (_Float16*)carve((size_t)L_ * D_ * D_ * 2);
  _Float16* W1H   = (_Float16*)carve((size_t)L_ * DFF_ * D_ * 2);
  _Float16* W2H   = (_Float16*)carve((size_t)L_ * D_ * DFF_ * 2);
  _Float16* headH = (_Float16*)carve((size_t)V_ * D_ * 2);
  float*    x     = (float*)carve((size_t)S_ * D_ * 4);
  float*    y     = (float*)carve((size_t)S_ * D_ * 4);
  _Float16* xh    = (_Float16*)carve((size_t)S_ * D_ * 2);
  _Float16* qh    = (_Float16*)carve((size_t)S_ * D_ * 2);
  _Float16* kh    = (_Float16*)carve((size_t)S_ * D_ * 2);
  _Float16* vh    = (_Float16*)carve((size_t)S_ * D_ * 2);
  _Float16* vt    = (_Float16*)carve((size_t)D_ * S_ * 2);   // V^T [D, S]
  _Float16* ah    = (_Float16*)carve((size_t)S_ * D_ * 2);
  _Float16* hh    = (_Float16*)carve((size_t)S_ * DFF_ * 2);

  auto cvt = [&](const float* src, _Float16* dst, long long n) {
    long long b = (n + 255) / 256;
    int blocks = (int)(b > 4096 ? 4096 : b);
    cvt_f32_f16_kernel<<<blocks, 256, 0, stream>>>(src, dst, n);
  };
  cvt(Wq, WqH, (long long)L_ * D_ * D_);
  cvt(Wk, WkH, (long long)L_ * D_ * D_);
  cvt(Wv, WvH, (long long)L_ * D_ * D_);
  cvt(Wo, WoH, (long long)L_ * D_ * D_);
  cvt(W1, W1H, (long long)L_ * DFF_ * D_);
  cvt(W2, W2H, (long long)L_ * D_ * DFF_);
  cvt(head_w, headH, (long long)V_ * D_);

  embed_kernel<<<S_, 256, 0, stream>>>(ids, emb, x, xh);

  const dim3 gDD(D_ / 64, S_ / 128);    // N=512 GEMMs
  const dim3 gDF(DFF_ / 64, S_ / 128);  // N=2048 GEMM (FF1)
  const dim3 gHD(V_ / 64, S_ / 128);    // head GEMM

  for (int l = 0; l < L_; ++l) {
    const size_t wo  = (size_t)l * D_ * D_;
    const size_t wf1 = (size_t)l * DFF_ * D_;
    const size_t wf2 = (size_t)l * D_ * DFF_;

    gemm_f16_kernel<0><<<gDD, 256, 0, stream>>>(xh, WqH + wo, bq + l * D_,
                                                nullptr, nullptr, qh, S_, D_, D_);
    gemm_f16_kernel<0><<<gDD, 256, 0, stream>>>(xh, WkH + wo, bk + l * D_,
                                                nullptr, nullptr, kh, S_, D_, D_);
    gemm_f16_kernel<0><<<gDD, 256, 0, stream>>>(xh, WvH + wo, bv + l * D_,
                                                nullptr, nullptr, vh, S_, D_, D_);

    transpose_f16_kernel<<<dim3(D_ / 32, S_ / 32), 256, 0, stream>>>(vh, vt);

    flash_attn_kernel<<<dim3(S_ / 16, H_), 32, 0, stream>>>(qh, kh, vt, amask, ah);

    gemm_f16_kernel<0><<<gDD, 256, 0, stream>>>(ah, WoH + wo, bo + l * D_,
                                                x, y, nullptr, S_, D_, D_);
    rmsnorm_kernel<<<S_, 128, 0, stream>>>(y, g1 + l * D_, x, xh);

    gemm_f16_kernel<1><<<gDF, 256, 0, stream>>>(xh, W1H + wf1, b1 + l * DFF_,
                                                nullptr, nullptr, hh, S_, DFF_, D_);
    gemm_f16_kernel<0><<<gDD, 256, 0, stream>>>(hh, W2H + wf2, b2 + l * D_,
                                                x, y, nullptr, S_, D_, DFF_);
    rmsnorm_kernel<<<S_, 128, 0, stream>>>(y, g2 + l * D_, x, xh);
  }

  gemm_f16_kernel<0><<<gHD, 256, 0, stream>>>(xh, headH, head_b,
                                              nullptr, out, nullptr, S_, V_, D_);
}
